// TransformerDecoder_73452530696519
// MI455X (gfx1250) — compile-verified, matrix operand order checked by reading, not run on hardware
//
#include <hip/hip_runtime.h>

// ---------------------------------------------------------------------------
// Problem constants (match reference)
// ---------------------------------------------------------------------------
constexpr int kB  = 2;
constexpr int kT  = 1024;
constexpr int kC  = 1024;
constexpr int kH  = 8;
constexpr int kHD = kC / kH;     // 128
constexpr int kL  = 4;
constexpr int kFF = 100;
constexpr int kFFp = 128;        // FFN hidden padded to a clean WMMA K
constexpr int kV  = 32000;
constexpr int kM  = kB * kT;     // 2048 activation rows
constexpr float kEPS   = 1e-5f;
constexpr float kSCALE = 0.08838834764831845f;  // 1/sqrt(128)

typedef __attribute__((ext_vector_type(16))) _Float16 v16h;
typedef __attribute__((ext_vector_type(8)))  _Float16 v8h;
typedef __attribute__((ext_vector_type(8)))  float    v8f;

// Assemble a 16x32 f16 WMMA fragment for this lane from two contiguous
// 8-half (16-byte) chunks.  Per ISA 7.12.2, lane-group grp (lane>>4) holds
// K = kbase..kbase+7 in elements 0..7 and K = kbase+16..kbase+23 in 8..15,
// with kbase = k0 + 8*grp.
__device__ __forceinline__ v16h load_frag(const _Float16* __restrict__ p, int kbase) {
    v8h lo = *(const v8h*)(p + kbase);
    v8h hi = *(const v8h*)(p + kbase + 16);
    return __builtin_shufflevector(lo, hi, 0, 1, 2, 3, 4, 5, 6, 7,
                                           8, 9, 10, 11, 12, 13, 14, 15);
}

// ---------------------------------------------------------------------------
// Embedding gather: h[m, :] = emb[x[m], :]
// ---------------------------------------------------------------------------
__global__ void embed_kernel(const int* __restrict__ x,
                             const float* __restrict__ emb,
                             float* __restrict__ h) {
    const size_t m = blockIdx.x;
    const int tok  = x[m];
    const float* src = emb + (size_t)tok * kC;
    float* dst = h + m * kC;
    for (int c = threadIdx.x; c < kC; c += blockDim.x) dst[c] = src[c];
}

// ---------------------------------------------------------------------------
// LayerNorm over C=1024 -> f16, one 256-thread block per row (LDS reduce).
// ---------------------------------------------------------------------------
__global__ void layernorm_f16_kernel(const float* __restrict__ xin,
                                     const float* __restrict__ g,
                                     const float* __restrict__ b,
                                     _Float16* __restrict__ out) {
    __shared__ float s_sum[256];
    __shared__ float s_sq[256];
    const size_t row = blockIdx.x;
    const float* p = xin + row * kC;
    const int t = threadIdx.x;

    float a = 0.f, q = 0.f;
    for (int c = t; c < kC; c += blockDim.x) {
        float v = p[c];
        a += v; q += v * v;
    }
    s_sum[t] = a; s_sq[t] = q;
    __syncthreads();
    for (int s = 128; s > 0; s >>= 1) {
        if (t < s) { s_sum[t] += s_sum[t + s]; s_sq[t] += s_sq[t + s]; }
        __syncthreads();
    }
    const float mean = s_sum[0] * (1.0f / kC);
    const float var  = s_sq[0] * (1.0f / kC) - mean * mean;
    const float inv  = rsqrtf(var + kEPS);

    _Float16* o = out + row * kC;
    for (int c = t; c < kC; c += blockDim.x)
        o[c] = (_Float16)((p[c] - mean) * inv * g[c] + b[c]);
}

// ---------------------------------------------------------------------------
// Weight pack: f32 [Kin, Nin] row-major  ->  f16 transposed [Np, Kp] row-major
// (zero-padded).  32x32 LDS tile transpose, 256 threads/block.
// Grid: (Kp/32, Np/32).
// ---------------------------------------------------------------------------
__global__ __launch_bounds__(256)
void pack_wT_kernel(const float* __restrict__ W, int Kin, int Nin,
                    _Float16* __restrict__ Wt, int Kp) {
    __shared__ _Float16 tile[32][33];
    const int k0 = blockIdx.x * 32;
    const int n0 = blockIdx.y * 32;
    const int tx = threadIdx.x & 31;
    const int ty = threadIdx.x >> 5;  // 0..7
#pragma unroll
    for (int it = 0; it < 4; ++it) {
        const int k = k0 + ty + it * 8;
        const int n = n0 + tx;
        const float v = (k < Kin && n < Nin) ? W[(size_t)k * Nin + n] : 0.f;
        tile[ty + it * 8][tx] = (_Float16)v;
    }
    __syncthreads();
#pragma unroll
    for (int it = 0; it < 4; ++it) {
        const int n = n0 + ty + it * 8;
        const int k = k0 + tx;
        Wt[(size_t)n * Kp + k] = tile[tx][ty + it * 8];
    }
}

// ---------------------------------------------------------------------------
// V transpose: v16 [B*T, C] (head-interleaved) -> vT [B,H,HD,T] f16
// Grid: (T/32, HD/32, B*H).
// ---------------------------------------------------------------------------
__global__ __launch_bounds__(256)
void transpose_v_kernel(const _Float16* __restrict__ v16,
                        _Float16* __restrict__ vT) {
    __shared__ _Float16 tile[32][33];
    const int bh = blockIdx.z;
    const int b = bh / kH;
    const int h = bh % kH;
    const int t0 = blockIdx.x * 32;
    const int d0 = blockIdx.y * 32;
    const int tx = threadIdx.x & 31;
    const int ty = threadIdx.x >> 5;
#pragma unroll
    for (int it = 0; it < 4; ++it) {
        const int tt = t0 + ty + it * 8;
        tile[ty + it * 8][tx] = v16[(size_t)(b * kT + tt) * kC + h * kHD + d0 + tx];
    }
    __syncthreads();
#pragma unroll
    for (int it = 0; it < 4; ++it) {
        const int d = d0 + ty + it * 8;
        vT[((size_t)bh * kHD + d) * kT + t0 + tx] = tile[tx][ty + it * 8];
    }
}

// ---------------------------------------------------------------------------
// Fast WMMA GEMM:  D[M,N] = A_f16[M,K] * Bt_f16[N,K]^T + bias
// All of M, K multiples of 32; Bt rows padded so every load is in range.
// Only the epilogue guards n < Nstore.  Per wave: one 16x16 tile.
// Block = 256 threads = 8 waves = 32(m) x 64(n) tile.
// Inner loop: 4x global_load_b128 + 1x v_wmma_f32_16x16x32_f16.
// ---------------------------------------------------------------------------
__global__ __launch_bounds__(256)
void gemm_f16t_kernel(const _Float16* __restrict__ A, int lda,
                      const _Float16* __restrict__ Bt, int ldb,
                      const float* __restrict__ bias, int Nbias,
                      float* __restrict__ Cout,
                      _Float16* __restrict__ C16, int ldc,
                      int Nstore, int Kdim, int relu, int accum) {
    const int lane = threadIdx.x & 31;
    const int wave = threadIdx.x >> 5;
    const int m0 = blockIdx.y * 32 + (wave >> 2) * 16;
    const int n0 = blockIdx.x * 64 + (wave & 3) * 16;
    const int grp = lane >> 4;
    const int r   = lane & 15;

    const _Float16* Ap = A  + (size_t)(m0 + r) * lda;
    const _Float16* Bp = Bt + (size_t)(n0 + r) * ldb;

    v8f acc = {};
#pragma unroll 2
    for (int k0 = 0; k0 < Kdim; k0 += 32) {
        const int kb = k0 + grp * 8;
        __builtin_prefetch(Bp + kb + 64, 0, 1);   // speculative; OOB dropped by HW
        v16h a  = load_frag(Ap, kb);
        v16h bf = load_frag(Bp, kb);
        acc = __builtin_amdgcn_wmma_f32_16x16x32_f16(
            false, a, false, bf, (short)0, acc, false, false);
    }

#pragma unroll
    for (int j = 0; j < 8; ++j) {
        const int m = m0 + grp * 8 + j;
        const int n = n0 + r;
        if (n < Nstore) {
            float v = acc[j] + ((n < Nbias) ? bias[n] : 0.f);
            if (relu) v = fmaxf(v, 0.f);
            const size_t idx = (size_t)m * ldc + n;
            if (accum) v += Cout[idx];
            if (Cout) Cout[idx] = v;
            if (C16)  C16[idx] = (_Float16)v;
        }
    }
}

// ---------------------------------------------------------------------------
// Attention scores: S[bh,i,j] = (Q_bh[i,:] . K_bh[j,:]) * scale
//                              + slope_h*(j-i)  (j<=i), else -inf
// ---------------------------------------------------------------------------
__global__ __launch_bounds__(256)
void attn_scores_kernel(const _Float16* __restrict__ q16,
                        const _Float16* __restrict__ k16,
                        float* __restrict__ scores) {
    const int bh = blockIdx.z;
    const int b = bh / kH;
    const int h = bh % kH;
    const int lane = threadIdx.x & 31;
    const int wave = threadIdx.x >> 5;
    const int i0 = blockIdx.y * 32 + (wave >> 2) * 16;
    const int j0 = blockIdx.x * 64 + (wave & 3) * 16;
    const int grp = lane >> 4;
    const int r   = lane & 15;

    float* srow = scores + (size_t)bh * kT * kT;

    if (j0 > i0 + 15) {  // tile fully above the diagonal (wave-uniform branch)
#pragma unroll
        for (int j = 0; j < 8; ++j)
            srow[(size_t)(i0 + grp * 8 + j) * kT + (j0 + r)] = -__builtin_inff();
        return;
    }

    const _Float16* Qp = q16 + (size_t)(b * kT + i0 + r) * kC + h * kHD;
    const _Float16* Kp = k16 + (size_t)(b * kT + j0 + r) * kC + h * kHD;

    v8f acc = {};
#pragma unroll
    for (int k0 = 0; k0 < kHD; k0 += 32) {
        v16h a  = load_frag(Qp, k0 + grp * 8);
        v16h bf = load_frag(Kp, k0 + grp * 8);
        acc = __builtin_amdgcn_wmma_f32_16x16x32_f16(
            false, a, false, bf, (short)0, acc, false, false);
    }

    const float slope = exp2f(-(float)(h + 1));
#pragma unroll
    for (int j = 0; j < 8; ++j) {
        const int qi = i0 + grp * 8 + j;
        const int kj = j0 + r;
        const float v = (kj <= qi)
            ? (acc[j] * kSCALE + slope * (float)(kj - qi))
            : -__builtin_inff();
        srow[(size_t)qi * kT + kj] = v;
    }
}

// ---------------------------------------------------------------------------
// Row softmax over T=1024, in place (f32 = the attn-map output) + f16 copy.
// ---------------------------------------------------------------------------
__global__ void softmax_rows_kernel(float* __restrict__ attn,
                                    _Float16* __restrict__ attn16) {
    __shared__ float red[256];
    const size_t row = blockIdx.x;
    float* p = attn + row * kT;
    const int t = threadIdx.x;

    float mx = -__builtin_inff();
    for (int c = t; c < kT; c += blockDim.x) mx = fmaxf(mx, p[c]);
    red[t] = mx; __syncthreads();
    for (int s = 128; s > 0; s >>= 1) {
        if (t < s) red[t] = fmaxf(red[t], red[t + s]);
        __syncthreads();
    }
    const float m = red[0];
    __syncthreads();

    float sum = 0.f;
    for (int c = t; c < kT; c += blockDim.x) {
        const float e = expf(p[c] - m);   // masked (-inf) -> 0
        p[c] = e;
        sum += e;
    }
    red[t] = sum; __syncthreads();
    for (int s = 128; s > 0; s >>= 1) {
        if (t < s) red[t] += red[t + s];
        __syncthreads();
    }
    const float inv = 1.0f / red[0];

    _Float16* o = attn16 + row * kT;
    for (int c = t; c < kT; c += blockDim.x) {
        const float v = p[c] * inv;
        p[c] = v;
        o[c] = (_Float16)v;
    }
}

// ---------------------------------------------------------------------------
// AV GEMM: o[bh,i,d] = sum_k attn[bh,i,k] * vT[bh,d,k]   (K = T = 1024)
// Both operands contiguous in K -> pure b128 loads + WMMA.
// ---------------------------------------------------------------------------
__global__ __launch_bounds__(256)
void attn_av_kernel(const _Float16* __restrict__ attn16,
                    const _Float16* __restrict__ vT,
                    _Float16* __restrict__ o16) {
    const int bh = blockIdx.z;
    const int b = bh / kH;
    const int h = bh % kH;
    const int lane = threadIdx.x & 31;
    const int wave = threadIdx.x >> 5;
    const int i0 = blockIdx.y * 32 + (wave >> 2) * 16;
    const int n0 = blockIdx.x * 64 + (wave & 3) * 16;   // head-dim cols
    const int grp = lane >> 4;
    const int r   = lane & 15;

    const _Float16* Ap = attn16 + ((size_t)bh * kT + i0 + r) * kT;
    const _Float16* Bp = vT + ((size_t)bh * kHD + n0 + r) * kT;

    v8f acc = {};
#pragma unroll 2
    for (int k0 = 0; k0 < kT; k0 += 32) {
        const int kb = k0 + grp * 8;
        v16h a  = load_frag(Ap, kb);
        v16h bf = load_frag(Bp, kb);
        acc = __builtin_amdgcn_wmma_f32_16x16x32_f16(
            false, a, false, bf, (short)0, acc, false, false);
    }

#pragma unroll
    for (int j = 0; j < 8; ++j) {
        const int qi = i0 + grp * 8 + j;
        o16[(size_t)(b * kT + qi) * kC + h * kHD + n0 + r] = (_Float16)acc[j];
    }
}

// ---------------------------------------------------------------------------
// Host-side orchestration
// ---------------------------------------------------------------------------
extern "C" void kernel_launch(void* const* d_in, const int* in_sizes, int n_in,
                              void* d_out, int out_size, void* d_ws, size_t ws_size,
                              hipStream_t stream) {
    (void)in_sizes; (void)n_in; (void)out_size; (void)ws_size;

    const int*   x     = (const int*)d_in[0];
    const float* emb   = (const float*)d_in[1];
    const float* Wq    = (const float*)d_in[2];
    const float* bq    = (const float*)d_in[3];
    const float* Wk    = (const float*)d_in[4];
    const float* bk    = (const float*)d_in[5];
    const float* Wv    = (const float*)d_in[6];
    const float* bv    = (const float*)d_in[7];
    const float* Wo    = (const float*)d_in[8];
    const float* bo    = (const float*)d_in[9];
    const float* W1    = (const float*)d_in[10];
    const float* b1    = (const float*)d_in[11];
    const float* W2    = (const float*)d_in[12];
    const float* b2    = (const float*)d_in[13];
    const float* ln1_g = (const float*)d_in[14];
    const float* ln1_b = (const float*)d_in[15];
    const float* ln3_g = (const float*)d_in[16];
    const float* ln3_b = (const float*)d_in[17];
    const float* lnf_g = (const float*)d_in[18];
    const float* lnf_b = (const float*)d_in[19];
    const float* Wout  = (const float*)d_in[20];
    const float* bout  = (const float*)d_in[21];

    float* logits    = (float*)d_out;                  // [B,T,V]
    float* attn_base = logits + (size_t)kB * kT * kV;  // [L,B,H,T,T]

    // Workspace carve-up (~130 MB)
    char* ws = (char*)d_ws;
    size_t off = 0;
    float*    h      = (float*)(ws + off);    off += (size_t)kM * kC * 4;
    _Float16* hn16   = (_Float16*)(ws + off); off += (size_t)kM * kC * 2;
    _Float16* q16    = (_Float16*)(ws + off); off += (size_t)kM * kC * 2;
    _Float16* k16    = (_Float16*)(ws + off); off += (size_t)kM * kC * 2;
    _Float16* v16    = (_Float16*)(ws + off); off += (size_t)kM * kC * 2;
    _Float16* o16    = (_Float16*)(ws + off); off += (size_t)kM * kC * 2;
    _Float16* vT16   = (_Float16*)(ws + off); off += (size_t)kB * kH * kHD * kT * 2;
    _Float16* ff16   = (_Float16*)(ws + off); off += (size_t)kM * kFFp * 2;
    _Float16* attn16 = (_Float16*)(ws + off); off += (size_t)kB * kH * kT * kT * 2;
    _Float16* wt16   = (_Float16*)(ws + off); off += (size_t)kV * kC * 2;  // packed-weight slab

    const dim3 blk(256);
    const dim3 gProj(kC / 64, kM / 32);               // 2048x1024 GEMMs
    const dim3 gScores(kT / 64, kT / 32, kB * kH);
    const dim3 gAV(kHD / 64, kT / 32, kB * kH);
    const dim3 gFF1(kFFp / 64, kM / 32);              // stores full padded 128 cols
    const dim3 gFF2(kC / 64, kM / 32);
    const dim3 gLogits(kV / 64, kM / 32);             // 500 x 64
    const dim3 gVT(kT / 32, kHD / 32, kB * kH);

    const dim3 gPackCC(kC / 32, kC / 32);             // [1024,1024] -> [1024,1024]
    const dim3 gPackW1(kC / 32, kFFp / 32);           // [1024,100]  -> [128,1024]
    const dim3 gPackW2(kFFp / 32, kC / 32);           // [100,1024]  -> [1024,128]
    const dim3 gPackOut(kC / 32, kV / 32);            // [1024,32000]-> [32000,1024]

    embed_kernel<<<kM, blk, 0, stream>>>(x, emb, h);

    for (int l = 0; l < kL; ++l) {
        const size_t wO = (size_t)l * kC * kC;
        const size_t vO = (size_t)l * kC;

        // ---- attention block ----
        layernorm_f16_kernel<<<kM, blk, 0, stream>>>(h, ln1_g + vO, ln1_b + vO, hn16);

        pack_wT_kernel<<<gPackCC, blk, 0, stream>>>(Wq + wO, kC, kC, wt16, kC);
        gemm_f16t_kernel<<<gProj, blk, 0, stream>>>(
            hn16, kC, wt16, kC, bq + vO, kC, nullptr, q16, kC, kC, kC, 0, 0);
        pack_wT_kernel<<<gPackCC, blk, 0, stream>>>(Wk + wO, kC, kC, wt16, kC);
        gemm_f16t_kernel<<<gProj, blk, 0, stream>>>(
            hn16, kC, wt16, kC, bk + vO, kC, nullptr, k16, kC, kC, kC, 0, 0);
        pack_wT_kernel<<<gPackCC, blk, 0, stream>>>(Wv + wO, kC, kC, wt16, kC);
        gemm_f16t_kernel<<<gProj, blk, 0, stream>>>(
            hn16, kC, wt16, kC, bv + vO, kC, nullptr, v16, kC, kC, kC, 0, 0);

        float* attn_l = attn_base + (size_t)l * kB * kH * kT * kT;
        attn_scores_kernel<<<gScores, blk, 0, stream>>>(q16, k16, attn_l);
        softmax_rows_kernel<<<kB * kH * kT, blk, 0, stream>>>(attn_l, attn16);
        transpose_v_kernel<<<gVT, blk, 0, stream>>>(v16, vT16);
        attn_av_kernel<<<gAV, blk, 0, stream>>>(attn16, vT16, o16);

        pack_wT_kernel<<<gPackCC, blk, 0, stream>>>(Wo + wO, kC, kC, wt16, kC);
        gemm_f16t_kernel<<<gProj, blk, 0, stream>>>(        // h += o @ Wo + bo
            o16, kC, wt16, kC, bo + vO, kC, h, nullptr, kC, kC, kC, 0, 1);

        // ---- FFN block ----
        layernorm_f16_kernel<<<kM, blk, 0, stream>>>(h, ln3_g + vO, ln3_b + vO, hn16);

        // W1: [1024,100] -> padded [128,1024]; write all 128 cols (padding = relu(0)=0)
        pack_wT_kernel<<<gPackW1, blk, 0, stream>>>(
            W1 + (size_t)l * kC * kFF, kC, kFF, wt16, kC);
        gemm_f16t_kernel<<<gFF1, blk, 0, stream>>>(
            hn16, kC, wt16, kC, b1 + (size_t)l * kFF, kFF,
            nullptr, ff16, kFFp, kFFp, kC, 1, 0);

        // W2: [100,1024] -> padded [1024,128]
        pack_wT_kernel<<<gPackW2, blk, 0, stream>>>(
            W2 + (size_t)l * kFF * kC, kFF, kC, wt16, kFFp);
        gemm_f16t_kernel<<<gFF2, blk, 0, stream>>>(         // h += ff1 @ W2 + b2
            ff16, kFFp, wt16, kFFp, b2 + vO, kC, h, nullptr, kC, kC, kFFp, 0, 1);
    }

    // ---- final LN + logits ----
    layernorm_f16_kernel<<<kM, blk, 0, stream>>>(h, lnf_g, lnf_b, hn16);
    pack_wT_kernel<<<gPackOut, blk, 0, stream>>>(Wout, kC, kV, wt16, kC);
    gemm_f16t_kernel<<<gLogits, blk, 0, stream>>>(
        hn16, kC, wt16, kC, bout, kV, logits, nullptr, kV, kV, kC, 0, 0);
}
